// CausalMHAWithState_40561671143562
// MI455X (gfx1250) — matrix operand order repaired and checked by reading, not compiled, not run on hardware
//
#include <hip/hip_runtime.h>
#include <hip/hip_bf16.h>

typedef __bf16 bf16_t;
typedef __attribute__((ext_vector_type(16))) __bf16 v16bf;
typedef __attribute__((ext_vector_type(8)))  __bf16 v8bf;
typedef __attribute__((ext_vector_type(8)))  float  v8f;
typedef __attribute__((ext_vector_type(4)))  unsigned int u32x4;
typedef __attribute__((ext_vector_type(8)))  int i32x8;
typedef __attribute__((ext_vector_type(4)))  int i32x4;

#define SEQ    3072
#define DH     128
#define NBG    16          // b(2) * heads(8)
#define SBLK   64          // s rows per workgroup (4 waves x 16)
#define KCHUNK 32
#define WSZ    (8 * 128 * 1024)   // one transposed weight tensor, elements

static __device__ __forceinline__ v8f wmma_bf16(v16bf a, v16bf b, v8f c) {
    return __builtin_amdgcn_wmma_f32_16x16x32_bf16(
        false, a, false, b, (short)0, c, false, false);
}
static __device__ __forceinline__ v8bf ld8(const bf16_t* p) { return *(const v8bf*)p; }
static __device__ __forceinline__ v16bf mk16(v8bf lo, v8bf hi) {
    return __builtin_shufflevector(lo, hi, 0,1,2,3,4,5,6,7,8,9,10,11,12,13,14,15);
}
static __device__ __forceinline__ v8bf cvt8(float4 a, float4 b) {
    v8bf r;
    r[0] = (bf16_t)a.x; r[1] = (bf16_t)a.y; r[2] = (bf16_t)a.z; r[3] = (bf16_t)a.w;
    r[4] = (bf16_t)b.x; r[5] = (bf16_t)b.y; r[6] = (bf16_t)b.z; r[7] = (bf16_t)b.w;
    return r;
}
// single-instruction exp2 (v_exp_f32)
static __device__ __forceinline__ float fast_exp2(float x) {
#if __has_builtin(__builtin_amdgcn_exp2f)
    return __builtin_amdgcn_exp2f(x);
#else
    return exp2f(x);
#endif
}

// ---------------------------------------------------------------------------
// Tensor Data Mover: async copy of a 2D bf16 tile (row-major) global -> LDS.
// D# per CDNA5 ISA ch.8 (2D tensor, type=2, data_size=2B).
// ---------------------------------------------------------------------------
#if __has_builtin(__builtin_amdgcn_tensor_load_to_lds)
#define HAVE_TDM 1
#if __has_include(<hip/amd_detail/amd_gfx1250_TDM.h>)
#define TDM_6ARG 1
#endif
static __device__ __forceinline__ void tdm_load_2d(const void* gsrc, unsigned lds_off,
                                                   unsigned tile_w, unsigned tile_h,
                                                   unsigned row_stride)
{
    unsigned long long ga = (unsigned long long)(uintptr_t)gsrc;
    u32x4 g0;
    g0[0] = 1u;                                    // count=1, user descriptor
    g0[1] = lds_off;                               // lds_addr (bytes)
    g0[2] = (unsigned)ga;                          // global_addr[31:0]
    g0[3] = (unsigned)((ga >> 32) & 0x1FFFFFFu) | (2u << 30);  // [56:32] | type=2
    i32x8 g1;
    g1[0] = (1 << 16);                             // data_size=1 -> 2 bytes/elt
    g1[1] = (int)((row_stride & 0xFFFFu) << 16);   // tensor_dim0 lo   (bits 79:48)
    g1[2] = (int)((row_stride >> 16) | ((3072u & 0xFFFFu) << 16)); // dim0 hi|dim1 lo
    g1[3] = (int)((3072u >> 16) | (tile_w << 16)); // dim1 hi | tile_dim0
    g1[4] = (int)(tile_h);                         // tile_dim1, tile_dim2=0
    g1[5] = (int)row_stride;                       // tensor_dim0_stride
    g1[6] = 0;
    g1[7] = 0;
    i32x4 z4; z4[0] = z4[1] = z4[2] = z4[3] = 0;
#ifdef TDM_6ARG
    i32x8 z8;
    #pragma unroll
    for (int i = 0; i < 8; ++i) z8[i] = 0;
    __builtin_amdgcn_tensor_load_to_lds(g0, g1, z4, z4, z8, 0);
#else
    __builtin_amdgcn_tensor_load_to_lds(g0, g1, z4, z4, 0);
#endif
}
#else
#define HAVE_TDM 0
#endif

// ---------------------------------------------------------------------------
// Weight prep:  W[h][g][d][e] f32  ->  WT[g][e][h*128+d] bf16 (transposed).
// grid = (8 g, 8 h, 4 d-chunks), block = 128.
// ---------------------------------------------------------------------------
__global__ __launch_bounds__(128)
void wprep_kernel(const float* __restrict__ W, bf16_t* __restrict__ WT)
{
    const int g  = blockIdx.x;
    const int h  = blockIdx.y;
    const int d0 = blockIdx.z * 32;
    const int tid = threadIdx.x;

    __shared__ bf16_t wtile[DH][32];   // [e][d] 8 KB

    {   // coalesced f32 read + convert + LDS transpose
        const int e0   = (tid & 3) * 32;
        const int drow = tid >> 2;                 // 0..31
        const float* p = W + (((size_t)h * 8 + g) * 128 + (size_t)(d0 + drow)) * 128 + e0;
        #pragma unroll
        for (int j = 0; j < 32; j += 4) {
            float4 f = *(const float4*)(p + j);
            wtile[e0 + j + 0][drow] = (bf16_t)f.x;
            wtile[e0 + j + 1][drow] = (bf16_t)f.y;
            wtile[e0 + j + 2][drow] = (bf16_t)f.z;
            wtile[e0 + j + 3][drow] = (bf16_t)f.w;
        }
    }
    __syncthreads();
    bf16_t* dst = WT + ((size_t)g * 128 + tid) * 1024 + h * 128 + d0;
    *(v16bf*)(dst)      = *(const v16bf*)&wtile[tid][0];
    *(v16bf*)(dst + 16) = *(const v16bf*)&wtile[tid][16];
}

// ---------------------------------------------------------------------------
// Projection:  out[b,g,s,e] = sum_k x[b,s,k] * WT[g,e,k]   (bf16 out)
// Weight k-slices (128 e-rows x 64 k) staged by TDM into double-buffered LDS;
// DMA of slice cc+1 overlaps the 16 WMMAs of slice cc.  B-tile LDS loads are
// software-rotated one WMMA ahead.
// grid = (48, 16), block = 128 (4 waves; wave w -> 16 s-rows).
// ---------------------------------------------------------------------------
__global__ __launch_bounds__(128)
void proj_kernel(const float* __restrict__ x,
                 const bf16_t* __restrict__ wt_st,
                 const bf16_t* __restrict__ wt_mid,
                 const bf16_t* __restrict__ wt_en,
                 bf16_t* __restrict__ out)
{
    const int sblk = blockIdx.x;
    const int bg   = blockIdx.y;
    const int b    = bg >> 3;
    const int g    = bg & 7;
    const int tid  = threadIdx.x;
    const int wave = tid >> 5;
    const int lane = tid & 31;
    const int hl   = lane >> 4;
    const int lr   = lane & 15;
    const int sbase = sblk * SBLK;
    const int s0    = sbase + wave * 16;

    const bf16_t* WT = (sbase < 512) ? wt_st : ((sbase < 2560) ? wt_mid : wt_en);
    const bf16_t* wg = WT + (size_t)g * 128 * 1024;   // [e=128][k=1024]

    __shared__ bf16_t wbuf[2][DH][64];   // 32 KB, double-buffered weight slice

    v8f acc[8];
    #pragma unroll
    for (int i = 0; i < 8; ++i)
        #pragma unroll
        for (int j = 0; j < 8; ++j) acc[i][j] = 0.0f;

#if HAVE_TDM
    if (wave == 0)
        tdm_load_2d(wg, (unsigned)(uintptr_t)&wbuf[0][0][0],
                    /*w=*/64, /*h=*/DH, /*stride=*/1024);
#endif

    for (int cc = 0; cc < 16; ++cc) {              // K = 1024 in slices of 64
#if HAVE_TDM
        if (wave == 0) __builtin_amdgcn_s_wait_tensorcnt((short)0);
        __syncthreads();                           // slice cc visible to all
        if (wave == 0 && cc + 1 < 16)
            tdm_load_2d(wg + (cc + 1) * 64,
                        (unsigned)(uintptr_t)&wbuf[(cc + 1) & 1][0][0],
                        64, DH, 1024);
#else
        {   // fallback: manual cooperative staging (thread = e-row)
            const bf16_t* src = wg + (size_t)tid * 1024 + cc * 64;
            #pragma unroll
            for (int j = 0; j < 4; ++j)
                *(v16bf*)&wbuf[cc & 1][tid][j * 16] = *(const v16bf*)(src + j * 16);
        }
        __syncthreads();
#endif

        #pragma unroll
        for (int c2 = 0; c2 < 2; ++c2) {
            const int c  = cc * 2 + c2;            // 32-wide k chunk index
            const int h  = c >> 2;
            const int d0 = (c & 3) * 32;

            // A operand: 16 s-rows x 32 k (f32 -> bf16 on the fly)
            const float* xrow = x + (((size_t)b * 8 + h) * SEQ + (s0 + lr)) * DH;
            const int ka = d0 + hl * 16;
            float4 a0 = *(const float4*)(xrow + ka);
            float4 a1 = *(const float4*)(xrow + ka + 4);
            float4 a2 = *(const float4*)(xrow + ka + 8);
            float4 a3 = *(const float4*)(xrow + ka + 12);
            v16bf av = mk16(cvt8(a0, a1), cvt8(a2, a3));

            // B tiles from LDS, rotated one load ahead of the WMMA chain
            const bf16_t* bbase = &wbuf[cc & 1][lr][c2 * 32 + hl * 16];
            v16bf bcur = *(const v16bf*)bbase;
            #pragma unroll
            for (int et = 0; et < 8; ++et) {
                v16bf bnxt;
                if (et < 7) bnxt = *(const v16bf*)(bbase + (et + 1) * 16 * 64);
                acc[et] = wmma_bf16(av, bcur, acc[et]);
                bcur = bnxt;
            }
        }
    }

    bf16_t* obase = out + ((size_t)bg * SEQ) * DH;
    #pragma unroll
    for (int et = 0; et < 8; ++et)
        #pragma unroll
        for (int r = 0; r < 8; ++r)
            obase[(size_t)(s0 + r + hl * 8) * DH + et * 16 + lr] = (bf16_t)acc[et][r];
}

// ---------------------------------------------------------------------------
// RoPE (in place on bf16 Q and K).  grid = (3072, 16, 2), block = 64.
// ---------------------------------------------------------------------------
__global__ void rope_kernel(bf16_t* __restrict__ q, bf16_t* __restrict__ k,
                            const int* __restrict__ off)
{
    const int s  = blockIdx.x;
    const int bg = blockIdx.y;
    const int i  = threadIdx.x;            // 0..63
    bf16_t* t = (blockIdx.z == 0 ? q : k) + ((size_t)bg * SEQ + s) * DH;

    const float inv = __expf(-9.2103403719761836f * ((float)(2 * i) / 128.0f));
    const float ang = (float)(s + off[0]) * inv;
    const float c = __cosf(ang), sn = __sinf(ang);
    const float t1 = (float)t[i], t2 = (float)t[i + 64];
    t[i]      = (bf16_t)(t1 * c - t2 * sn);
    t[i + 64] = (bf16_t)(t2 * c + t1 * sn);
}

// ---------------------------------------------------------------------------
// V transpose:  V[bh][s][d] -> VT[bh][d][s]  (bf16), LDS-tiled.
// grid = (16, 96), block = 128.
// ---------------------------------------------------------------------------
__global__ __launch_bounds__(128)
void vtrans_kernel(const bf16_t* __restrict__ V, bf16_t* __restrict__ VT)
{
    const int bg = blockIdx.x;
    const int sb = blockIdx.y * 32;
    const int tid = threadIdx.x;

    __shared__ bf16_t vtile[DH][32];   // [d][s] 8 KB

    {
        const int srow = tid & 31;
        const int dd0  = (tid >> 5) * 32;
        const bf16_t* p = V + ((size_t)bg * SEQ + sb + srow) * DH + dd0;
        #pragma unroll
        for (int j = 0; j < 32; j += 8) {
            v8bf v = ld8(p + j);
            #pragma unroll
            for (int u = 0; u < 8; ++u) vtile[dd0 + j + u][srow] = v[u];
        }
    }
    __syncthreads();
    bf16_t* dst = VT + ((size_t)bg * DH + tid) * SEQ + sb;   // row d = tid
    *(v16bf*)(dst)      = *(const v16bf*)&vtile[tid][0];
    *(v16bf*)(dst + 16) = *(const v16bf*)&vtile[tid][16];
}

// ---------------------------------------------------------------------------
// Flash attention. K tile (32x128) and VT tile (128x32) both staged by TDM
// into double-buffered LDS; online softmax runs in the log2 domain so every
// exponential is a single v_exp_f32.
// grid = (48, 16), block = 128 (4 waves; wave -> 16 q-rows).
// ---------------------------------------------------------------------------
__global__ __launch_bounds__(128)
void attn_kernel(const bf16_t* __restrict__ Q, const bf16_t* __restrict__ K,
                 const bf16_t* __restrict__ VT, float* __restrict__ out)
{
    const int qblk = blockIdx.x;
    const int bh   = blockIdx.y;
    const int tid  = threadIdx.x;
    const int wave = tid >> 5;
    const int lane = tid & 31;
    const int hl   = lane >> 4;
    const int lr   = lane & 15;
    const size_t base = (size_t)bh * SEQ * DH;
    const int q0 = qblk * SBLK + wave * 16;

    __shared__ bf16_t kbuf[2][KCHUNK][DH];    // 16 KB, K rows (key x d)
#if HAVE_TDM
    __shared__ bf16_t vbuf[2][DH][KCHUNK];    // 16 KB, VT rows (d x key)
#endif
    __shared__ bf16_t pbuf[4][16][KCHUNK];    // per-wave P tile, 4 KB

    const bf16_t* vtb = VT + base;            // [d][s]

    // preload Q A-operand tiles
    const bf16_t* qrow = Q + base + (size_t)(q0 + lr) * DH;
    v16bf qa[4];
    #pragma unroll
    for (int c = 0; c < 4; ++c) {
        const bf16_t* p = qrow + c * 32 + hl * 16;
        qa[c] = mk16(ld8(p), ld8(p + 8));
    }

    v8f acc[8];
    float mi[8], li[8];
    #pragma unroll
    for (int i = 0; i < 8; ++i) {
        mi[i] = -INFINITY; li[i] = 0.0f;
        #pragma unroll
        for (int j = 0; j < 8; ++j) acc[i][j] = 0.0f;
    }

    const int nchunks = qblk * 2 + 2;

#if HAVE_TDM
    if (wave == 0) {
        tdm_load_2d(K + base, (unsigned)(uintptr_t)&kbuf[0][0][0],
                    /*w=*/DH, /*h=*/KCHUNK, /*stride=*/DH);
        tdm_load_2d(vtb, (unsigned)(uintptr_t)&vbuf[0][0][0],
                    /*w=*/KCHUNK, /*h=*/DH, /*stride=*/SEQ);
    }
#endif

    for (int kc = 0; kc < nchunks; ++kc) {
        const int kb = kc * KCHUNK;

#if HAVE_TDM
        if (wave == 0) __builtin_amdgcn_s_wait_tensorcnt((short)0);
        __syncthreads();                      // tiles[kc] visible to all waves
        if (wave == 0 && kc + 1 < nchunks) {  // prefetch next tiles async
            tdm_load_2d(K + base + (size_t)(kb + KCHUNK) * DH,
                        (unsigned)(uintptr_t)&kbuf[(kc + 1) & 1][0][0],
                        DH, KCHUNK, DH);
            tdm_load_2d(vtb + kb + KCHUNK,
                        (unsigned)(uintptr_t)&vbuf[(kc + 1) & 1][0][0],
                        KCHUNK, DH, SEQ);
        }
#else
        {   // fallback: manual cooperative K staging
            const int key = tid & 31;
            const int dd0 = (tid >> 5) * 32;
            const bf16_t* src = K + base + (size_t)(kb + key) * DH + dd0;
            *(v16bf*)&kbuf[kc & 1][key][dd0]      = *(const v16bf*)src;
            *(v16bf*)&kbuf[kc & 1][key][dd0 + 16] = *(const v16bf*)(src + 16);
        }
        __syncthreads();
#endif

        if (kb <= q0 + 15) {
            // ---- S = Q K^T, two 16-key subtiles; LDS loads rotated 1 ahead
            const bf16_t* kbase = &kbuf[kc & 1][lr][hl * 16];
            v8f sv[2];
            {
                v8f s0w, s1w;
                #pragma unroll
                for (int j = 0; j < 8; ++j) { s0w[j] = 0.0f; s1w[j] = 0.0f; }
                v16bf kcur = *(const v16bf*)kbase;
                #pragma unroll
                for (int i = 0; i < 8; ++i) {       // i = t*4 + c
                    v16bf knxt;
                    if (i < 7) {
                        const int t = (i + 1) >> 2, c = (i + 1) & 3;
                        knxt = *(const v16bf*)(kbase + t * 16 * DH + c * 32);
                    }
                    if (i < 4) s0w = wmma_bf16(qa[i & 3], kcur, s0w);
                    else       s1w = wmma_bf16(qa[i & 3], kcur, s1w);
                    kcur = knxt;
                }
                sv[0] = s0w; sv[1] = s1w;
            }
            // log2-domain scale: 1/sqrt(128) * log2(e)
            const float scl2 = 0.08838834764831845f * 1.4426950408889634f;
            #pragma unroll
            for (int t = 0; t < 2; ++t) {
                const int key = kb + t * 16 + lr;
                #pragma unroll
                for (int r = 0; r < 8; ++r) {
                    const int qi = q0 + r + hl * 8;
                    float v = sv[t][r] * scl2;
                    sv[t][r] = (key > qi) ? -INFINITY : v;
                }
            }
            // ---- row max across the 16 lanes sharing these rows
            float mt[8];
            #pragma unroll
            for (int r = 0; r < 8; ++r) mt[r] = fmaxf(sv[0][r], sv[1][r]);
            #pragma unroll
            for (int m = 1; m < 16; m <<= 1)
                #pragma unroll
                for (int r = 0; r < 8; ++r)
                    mt[r] = fmaxf(mt[r], __shfl_xor(mt[r], m, 32));

            // ---- online softmax update (log2 domain, exp2 = 1 HW op)
            float scale[8], rs[8];
            #pragma unroll
            for (int r = 0; r < 8; ++r) {
                const float mn = fmaxf(mi[r], mt[r]);
                scale[r] = (mi[r] == -INFINITY) ? 0.0f : fast_exp2(mi[r] - mn);
                const float safem = (mn == -INFINITY) ? 0.0f : mn;
                const float p0 = fast_exp2(sv[0][r] - safem);
                const float p1 = fast_exp2(sv[1][r] - safem);
                sv[0][r] = p0; sv[1][r] = p1;
                rs[r] = p0 + p1;
                mi[r] = mn;
            }
            #pragma unroll
            for (int m = 1; m < 16; m <<= 1)
                #pragma unroll
                for (int r = 0; r < 8; ++r)
                    rs[r] += __shfl_xor(rs[r], m, 32);
            #pragma unroll
            for (int r = 0; r < 8; ++r)
                li[r] = li[r] * scale[r] + rs[r];
            #pragma unroll
            for (int dt = 0; dt < 8; ++dt)
                #pragma unroll
                for (int r = 0; r < 8; ++r)
                    acc[dt][r] *= scale[r];

            // ---- P: C-layout -> LDS -> A-layout (same-wave DS is in-order)
            #pragma unroll
            for (int t = 0; t < 2; ++t)
                #pragma unroll
                for (int r = 0; r < 8; ++r)
                    pbuf[wave][r + hl * 8][t * 16 + lr] = (bf16_t)sv[t][r];
            asm volatile("s_wait_dscnt 0" ::: "memory");

            const bf16_t* pr = &pbuf[wave][lr][0];
            v16bf pa = mk16(*(const v8bf*)(pr + hl * 16),
                            *(const v8bf*)(pr + 8 + hl * 16));

            // ---- O += P V, LDS loads rotated one WMMA ahead
#if HAVE_TDM
            const bf16_t* vbase = &vbuf[kc & 1][lr][hl * 16];
            const size_t vstep = 16 * KCHUNK;     // 16 d-rows per tile
#else
            const bf16_t* vbase = vtb + (size_t)lr * SEQ + kb + hl * 16;
            const size_t vstep = (size_t)16 * SEQ;
#endif
            v16bf vcur = *(const v16bf*)vbase;
            #pragma unroll
            for (int dt = 0; dt < 8; ++dt) {
                v16bf vnxt;
                if (dt < 7) vnxt = *(const v16bf*)(vbase + (dt + 1) * vstep);
                acc[dt] = wmma_bf16(pa, vcur, acc[dt]);
                vcur = vnxt;
            }
        }
    }

    float invl[8];
    #pragma unroll
    for (int r = 0; r < 8; ++r) invl[r] = 1.0f / li[r];
    #pragma unroll
    for (int dt = 0; dt < 8; ++dt)
        #pragma unroll
        for (int r = 0; r < 8; ++r)
            out[base + (size_t)(q0 + r + hl * 8) * DH + dt * 16 + lr] =
                acc[dt][r] * invl[r];
}

// ---------------------------------------------------------------------------
extern "C" void kernel_launch(void* const* d_in, const int* in_sizes, int n_in,
                              void* d_out, int out_size, void* d_ws, size_t ws_size,
                              hipStream_t stream)
{
    const float* x   = (const float*)d_in[0];
    const int*   off = (const int*)d_in[10];

    const size_t n = (size_t)2 * 8 * SEQ * DH;     // per-tensor elements
    bf16_t* Qbf = (bf16_t*)d_ws;
    bf16_t* Kbf = Qbf + n;
    bf16_t* Vbf = Kbf + n;
    bf16_t* WT  = Vbf + n;          // 9 * WSZ bf16
    bf16_t* VT  = WT;               // aliases WT (weights dead after proj)

    // weight prep: d_in[1..9] = Wq Wk Wv Wqs Wks Wvs Wqe Wke Wve
    for (int w = 0; w < 9; ++w)
        wprep_kernel<<<dim3(8, 8, 4), 128, 0, stream>>>(
            (const float*)d_in[1 + w], WT + (size_t)w * WSZ);

    dim3 gp(SEQ / SBLK, NBG), bp(128);
    proj_kernel<<<gp, bp, 0, stream>>>(x, WT + 3ull*WSZ, WT + 0ull*WSZ, WT + 6ull*WSZ, Qbf);
    proj_kernel<<<gp, bp, 0, stream>>>(x, WT + 4ull*WSZ, WT + 1ull*WSZ, WT + 7ull*WSZ, Kbf);
    proj_kernel<<<gp, bp, 0, stream>>>(x, WT + 5ull*WSZ, WT + 2ull*WSZ, WT + 8ull*WSZ, Vbf);

    rope_kernel<<<dim3(SEQ, NBG, 2), 64, 0, stream>>>(Qbf, Kbf, off);

    vtrans_kernel<<<dim3(NBG, SEQ / 32), 128, 0, stream>>>(Vbf, VT);

    attn_kernel<<<dim3(SEQ / SBLK, NBG), 128, 0, stream>>>(Qbf, Kbf, VT,
                                                           (float*)d_out);
}
